// SparsePillarSelfAttention_23072564314131
// MI455X (gfx1250) — matrix operand order repaired and checked by reading, not compiled, not run on hardware
//
#include <hip/hip_runtime.h>

// ---------------- problem constants ----------------
#define BB   2
#define NPB  8192
#define CC   128
#define HH   4
#define DH   32
#define KK   4096
#define LN_EPS 1e-5f

typedef __attribute__((ext_vector_type(4)))  _Float16 v4h;
typedef __attribute__((ext_vector_type(8)))  _Float16 v8h;
typedef __attribute__((ext_vector_type(16))) _Float16 v16h;
typedef __attribute__((ext_vector_type(8)))  float    v8f;

// Fragment layouts (wave32, 16x16x32 f16 WMMA):
//  A (16x32, MxK): lane = (row&15) + 16*((k&31)>>3 & 1)... precisely:
//      lane = row + 16*((k32>>3)&1), elem = (k32&7) | ((k32>>4)<<3)
//  B (32x16, KxN): lane = n + 16*((k32>>3)&1), elem = (k32&7) | ((k32>>4)<<3)
//  C (16x16, f32): lane = n + 16*(row>>3), elem v = row & 7

__device__ __forceinline__ unsigned lds_addr_of(const void* p) {
    return (unsigned)(unsigned long long)p;   // low 32 bits = LDS offset
}

__device__ __forceinline__ void async_b128x2(unsigned lds, const void* g) {
    asm volatile(
        "global_load_async_to_lds_b128 %0, %1, off\n\t"
        "global_load_async_to_lds_b128 %0, %1, off offset:16"
        :: "v"(lds), "v"(g) : "memory");
}

// =====================================================================
// Kernel 0: pass-through copy feats -> out
// =====================================================================
__global__ void copy_kernel(const float4* __restrict__ in, float4* __restrict__ out, int n4) {
    int i = blockIdx.x * blockDim.x + threadIdx.x;
    if (i < n4) out[i] = in[i];
}

// =====================================================================
// Kernel 1: per-batch squared-norm + top-K via LDS bitonic sort.
// =====================================================================
__global__ __launch_bounds__(1024)
void topk_kernel(const float* __restrict__ feats, int* __restrict__ idx_out) {
    __shared__ unsigned long long keys[NPB];   // 64 KB
    const int b = blockIdx.x;
    const int tid = threadIdx.x;
    const float* fb = feats + (size_t)b * NPB * CC;

    for (int r = tid; r < NPB; r += 1024) {
        const float4* row = (const float4*)(fb + (size_t)r * CC);
        float s = 0.f;
        #pragma unroll
        for (int i = 0; i < CC / 4; ++i) {
            float4 v = row[i];
            s += v.x * v.x + v.y * v.y + v.z * v.z + v.w * v.w;
        }
        unsigned int kb = __float_as_uint(s);
        keys[r] = ((unsigned long long)kb << 32) |
                  (unsigned long long)(unsigned int)(NPB - 1 - r);
    }
    __syncthreads();

    for (int size = 2; size <= NPB; size <<= 1) {
        for (int stride = size >> 1; stride > 0; stride >>= 1) {
            for (int t = tid; t < NPB / 2; t += 1024) {
                int i = ((t / stride) * (stride << 1)) + (t % stride);
                int j = i + stride;
                bool desc = ((i & size) == 0);
                unsigned long long a = keys[i], c = keys[j];
                bool sw = desc ? (a < c) : (a > c);
                if (sw) { keys[i] = c; keys[j] = a; }
            }
            __syncthreads();
        }
    }
    for (int t = tid; t < KK; t += 1024) {
        unsigned int low = (unsigned int)(keys[t] & 0xffffffffu);
        idx_out[b * KK + t] = NPB - 1 - (int)low;
    }
}

// =====================================================================
// Kernel 2: gather + QKV projection with fragment-order LDS staging.
// Grid (KK/64, 384/64, B); block 128 (4 waves).
// Outputs written directly in the attention kernel's fragment layouts:
//   qf: [B][H][K/16][lane 32][elem 16]  A-frag order, pre-scaled 1/sqrt(Dh)
//   kf: [B][H][K/16][lane 32][elem 16]  B-frag order for Q K^T
//   vf: [B][H][K/32][j 2][lane 32][elem 16]  B-frag order for P V
// =====================================================================
__global__ __launch_bounds__(128)
void qkv_kernel(const float* __restrict__ feats, const int* __restrict__ idxp,
                const float* __restrict__ w_in, const float* __restrict__ b_in,
                _Float16* __restrict__ qf, _Float16* __restrict__ kf,
                _Float16* __restrict__ vf) {
    __shared__ __align__(16) _Float16 Af[4][4][512];  // [wave][kstep] frag-order, 16 KB
    __shared__ __align__(16) _Float16 Bf[4][4][512];  // [kstep][j]    frag-order, 16 KB
    __shared__ int gidx[64];

    const int m0 = blockIdx.x * 64;
    const int n0 = blockIdx.y * 64;
    const int b  = blockIdx.z;
    const int tid = threadIdx.x;
    const int w = tid >> 5, lane = tid & 31, l16 = lane & 15, hi = lane >> 4;

    if (tid < 64) gidx[tid] = idxp[b * KK + m0 + tid];
    __syncthreads();

    // stage: quad-packed (4 consecutive k -> 4 consecutive frag elems, one lane)
    for (int e4 = tid; e4 < 64 * 32; e4 += 128) {
        int r = e4 >> 5;               // row (A) / out-col (B), 0..63
        int k = (e4 & 31) << 2;        // 0..124 step 4
        int kk = k >> 5, k32 = k & 31;

        float4 fa = *(const float4*)(feats + ((size_t)b * NPB + gidx[r]) * CC + k);
        {
            int lane_ = (r & 15) + (((k32 >> 3) & 1) << 4);
            int ep = (k32 & 7) | ((k32 >> 4) << 3);
            v4h h4; h4[0] = (_Float16)fa.x; h4[1] = (_Float16)fa.y;
                    h4[2] = (_Float16)fa.z; h4[3] = (_Float16)fa.w;
            *(v4h*)&Af[r >> 4][kk][lane_ * 16 + ep] = h4;
        }
        float4 fw = *(const float4*)(w_in + (size_t)(n0 + r) * CC + k);
        {
            int lane_ = (r & 15) + (((k32 >> 3) & 1) << 4);
            int ep = (k32 & 7) | ((k32 >> 4) << 3);
            v4h h4; h4[0] = (_Float16)fw.x; h4[1] = (_Float16)fw.y;
                    h4[2] = (_Float16)fw.z; h4[3] = (_Float16)fw.w;
            *(v4h*)&Bf[kk][r >> 4][lane_ * 16 + ep] = h4;
        }
    }
    __syncthreads();

    v8f acc[4] = {};
    #pragma unroll
    for (int kk = 0; kk < 4; ++kk) {
        v16h a = *(const v16h*)&Af[w][kk][lane * 16];
        #pragma unroll
        for (int j = 0; j < 4; ++j) {
            v16h bb = *(const v16h*)&Bf[kk][j][lane * 16];
            acc[j] = __builtin_amdgcn_wmma_f32_16x16x32_f16(
                false, a, false, bb, (short)0, acc[j], false, false);
        }
    }

    const int mat = n0 >> 7;  // 0=q, 1=k, 2=v (uniform per WG)
    #pragma unroll
    for (int j = 0; j < 4; ++j) {
        int gcol = n0 + j * 16 + l16;
        float bias = b_in[gcol];
        int rem = gcol & 127;
        int h = rem >> 5, d = rem & 31;
        if (mat == 0) {
            const float qs = 0.17677669529663687f;   // 1/sqrt(Dh)
            int tile = (m0 >> 4) + w;
            int ep = (d & 7) | ((d >> 4) << 3);
            int b3 = (d >> 3) & 1;
            _Float16* dst = qf + ((size_t)(b * HH + h) * (KK / 16) + tile) * 512;
            #pragma unroll
            for (int v = 0; v < 8; ++v)
                dst[(v + 8 * hi + 16 * b3) * 16 + ep] =
                    (_Float16)((acc[j][v] + bias) * qs);
        } else if (mat == 1) {
            int tile = (m0 >> 4) + w;
            int ep = (d & 7) | ((d >> 4) << 3);
            int b4 = (d >> 3) & 1;
            _Float16* dst = kf + ((size_t)(b * HH + h) * (KK / 16) + tile) * 512;
            #pragma unroll
            for (int v = 0; v < 8; ++v)
                dst[((v + 8 * hi) + 16 * b4) * 16 + ep] =
                    (_Float16)(acc[j][v] + bias);
        } else {
            // V: 8 consecutive frag elems -> one 16B store
            int kstep = (m0 >> 5) + (w >> 1);
            int jd = d >> 4;
            int lane_ = (d & 15) + 16 * hi;
            v8h pk;
            #pragma unroll
            for (int v = 0; v < 8; ++v) pk[v] = (_Float16)(acc[j][v] + bias);
            _Float16* dst = vf + ((size_t)(b * HH + h) * (KK / 32) + kstep) * 1024
                               + jd * 512;
            *(v8h*)(dst + lane_ * 16 + 8 * (w & 1)) = pk;
        }
    }
}

// =====================================================================
// Kernel 3: flash attention.  Grid (KK/64, H, B); block 128 (4 waves).
// K/V tiles (fragment-ordered, contiguous 4 KB) streamed into LDS via
// double-buffered global_load_async_to_lds_b128; softmax stats via
// __shfl_xor on C fragments; only the P transpose touches LDS.
// =====================================================================
__global__ __launch_bounds__(128)
void attn_kernel(const _Float16* __restrict__ qf, const _Float16* __restrict__ kf,
                 const _Float16* __restrict__ vf, _Float16* __restrict__ oh) {
    __shared__ __align__(16) _Float16 Kf[2][2048];      // [buf][j*512 + lane*16+e] 8 KB
    __shared__ __align__(16) _Float16 Vf[2][2048];      // [buf][(ks*2+j)*512 + ...] 8 KB
    __shared__ __align__(16) _Float16 Pf[4][2][512];    // [wave][ks] A-frag order  8 KB

    const int q0 = blockIdx.x * 64;
    const int h  = blockIdx.y;
    const int b  = blockIdx.z;
    const int tid = threadIdx.x;
    const int w = tid >> 5, lane = tid & 31, l16 = lane & 15, hi = lane >> 4;

    const size_t headq = (size_t)(b * HH + h) * (KK / 16) * 512;
    const _Float16* khead = kf + headq;                       // same geometry
    const _Float16* vhead = vf + (size_t)(b * HH + h) * (KK / 32) * 1024;

    // Q A-fragment straight from global (frag-ordered, 32B per lane)
    const v16h aq = *(const v16h*)(qf + headq + (size_t)(q0 / 16 + w) * 512 + lane * 16);

    v8f o0 = {}, o1 = {};
    v8f m_st, l_st = {};
    #pragma unroll
    for (int e = 0; e < 8; ++e) m_st[e] = -3.0e38f;

    const unsigned klds0 = lds_addr_of(&Kf[0][0]);
    const unsigned vlds0 = lds_addr_of(&Vf[0][0]);

    auto issue_tile = [&](int kt2, int buf2) {
        // K tile: 2048 halfs (4 KB); V tile: 2048 halfs; 32B per thread each
        async_b128x2(klds0 + buf2 * 4096 + tid * 32,
                     (const void*)((const char*)(khead + (size_t)kt2 * 2048) + tid * 32));
        async_b128x2(vlds0 + buf2 * 4096 + tid * 32,
                     (const void*)((const char*)(vhead + (size_t)kt2 * 2048) + tid * 32));
    };

    issue_tile(0, 0);
    for (int kt = 0; kt < KK / 64; ++kt) {
        const int buf = kt & 1;
        if (kt + 1 < KK / 64) {
            issue_tile(kt + 1, buf ^ 1);
            asm volatile("s_wait_asynccnt 0x4" ::: "memory");  // prev tile done
        } else {
            asm volatile("s_wait_asynccnt 0x0" ::: "memory");
        }
        __syncthreads();

        // S = (Q/sqrt(Dh)) K^T : 4 WMMA (Dh=32 is one K-step)
        v8f s[4];
        #pragma unroll
        for (int j = 0; j < 4; ++j) {
            v16h bk = *(const v16h*)&Kf[buf][j * 512 + lane * 16];
            v8f z = {};
            s[j] = __builtin_amdgcn_wmma_f32_16x16x32_f16(
                false, aq, false, bk, (short)0, z, false, false);
        }

        // row max over 64 cols: elementwise over j, then 16-lane xor reduce
        v8f mx = s[0];
        #pragma unroll
        for (int j = 1; j < 4; ++j)
            #pragma unroll
            for (int e = 0; e < 8; ++e) mx[e] = fmaxf(mx[e], s[j][e]);
        #pragma unroll
        for (int d2 = 1; d2 <= 8; d2 <<= 1)
            #pragma unroll
            for (int e = 0; e < 8; ++e)
                mx[e] = fmaxf(mx[e], __shfl_xor(mx[e], d2, 32));

        v8f mnew, corr;
        #pragma unroll
        for (int e = 0; e < 8; ++e) {
            mnew[e] = fmaxf(m_st[e], mx[e]);
            corr[e] = __expf(m_st[e] - mnew[e]);
            m_st[e] = mnew[e];
        }
        v8f rs = {};
        #pragma unroll
        for (int j = 0; j < 4; ++j)
            #pragma unroll
            for (int e = 0; e < 8; ++e) {
                float p = __expf(s[j][e] - mnew[e]);
                s[j][e] = p;
                rs[e] += p;
            }
        #pragma unroll
        for (int d2 = 1; d2 <= 8; d2 <<= 1)
            #pragma unroll
            for (int e = 0; e < 8; ++e) rs[e] += __shfl_xor(rs[e], d2, 32);
        #pragma unroll
        for (int e = 0; e < 8; ++e) {
            l_st[e] = l_st[e] * corr[e] + rs[e];
            o0[e] *= corr[e];
            o1[e] *= corr[e];
        }

        // P: C layout -> A-frag layout via LDS (per-wave region)
        #pragma unroll
        for (int j = 0; j < 4; ++j) {
            int col = j * 16 + l16;
            int ks = col >> 5, k32 = col & 31;
            int ep = (k32 & 7) | ((k32 >> 4) << 3);
            int b3 = (k32 >> 3) & 1;
            #pragma unroll
            for (int v = 0; v < 8; ++v)
                Pf[w][ks][(v + 8 * hi + 16 * b3) * 16 + ep] = (_Float16)s[j][v];
        }
        asm volatile("s_wait_dscnt 0x0" ::: "memory");  // wave-internal P visibility

        // O += P V : 4 WMMA
        #pragma unroll
        for (int ks = 0; ks < 2; ++ks) {
            v16h ap = *(const v16h*)&Pf[w][ks][lane * 16];
            #pragma unroll
            for (int j = 0; j < 2; ++j) {
                v16h bv = *(const v16h*)&Vf[buf][(ks * 2 + j) * 512 + lane * 16];
                if (j == 0)
                    o0 = __builtin_amdgcn_wmma_f32_16x16x32_f16(
                        false, ap, false, bv, (short)0, o0, false, false);
                else
                    o1 = __builtin_amdgcn_wmma_f32_16x16x32_f16(
                        false, ap, false, bv, (short)0, o1, false, false);
            }
        }
        __syncthreads();
    }

    // normalize, write O in [B][K][C] f16 (c = h*32 + d)
    #pragma unroll
    for (int v = 0; v < 8; ++v) {
        int m = q0 + 16 * w + v + 8 * hi;
        float inv = 1.f / l_st[v];
        int c0 = h * DH + l16;
        oh[((size_t)b * KK + m) * CC + c0]      = (_Float16)(o0[v] * inv);
        oh[((size_t)b * KK + m) * CC + c0 + 16] = (_Float16)(o1[v] * inv);
    }
}

// =====================================================================
// Kernel 4: out-projection + residual + LayerNorm + scatter.
// =====================================================================
__global__ __launch_bounds__(128)
void outproj_kernel(const _Float16* __restrict__ oh, const float* __restrict__ feats,
                    const int* __restrict__ idxp, const float* __restrict__ w_out,
                    const float* __restrict__ b_out, const float* __restrict__ lnw,
                    const float* __restrict__ lnb, float* __restrict__ out) {
    __shared__ float orow[CC];
    __shared__ float red[CC];
    const int t = blockIdx.x;
    const int b = t / KK, m = t % KK;
    const int c = threadIdx.x;
    const int g = idxp[b * KK + m];

    orow[c] = (float)oh[(size_t)(b * KK + m) * CC + c];
    __syncthreads();

    float acc = b_out[c];
    const float4* wr = (const float4*)(w_out + (size_t)c * CC);
    const float4* ov = (const float4*)orow;
    #pragma unroll 8
    for (int k = 0; k < CC / 4; ++k) {
        float4 wv = wr[k]; float4 o4 = ov[k];
        acc += wv.x * o4.x + wv.y * o4.y + wv.z * o4.z + wv.w * o4.w;
    }
    float xa = feats[((size_t)b * NPB + g) * CC + c];
    float hv = xa + acc;

    red[c] = hv;
    __syncthreads();
    for (int s = 64; s > 0; s >>= 1) { if (c < s) red[c] += red[c + s]; __syncthreads(); }
    float mu = red[0] * (1.f / CC);
    __syncthreads();
    float d = hv - mu;
    red[c] = d * d;
    __syncthreads();
    for (int s = 64; s > 0; s >>= 1) { if (c < s) red[c] += red[c + s]; __syncthreads(); }
    float var = red[0] * (1.f / CC);

    float y = d * rsqrtf(var + LN_EPS) * lnw[c] + lnb[c];
    out[((size_t)b * NPB + g) * CC + c] = y;
}

// =====================================================================
// launch
// =====================================================================
extern "C" void kernel_launch(void* const* d_in, const int* in_sizes, int n_in,
                              void* d_out, int out_size, void* d_ws, size_t ws_size,
                              hipStream_t stream) {
    (void)in_sizes; (void)n_in; (void)out_size; (void)ws_size;
    const float* feats = (const float*)d_in[0];
    // d_in[1] = batch_idx (contiguous equal groups) -> unused
    const float* in_w  = (const float*)d_in[2];
    const float* in_b  = (const float*)d_in[3];
    const float* out_w = (const float*)d_in[4];
    const float* out_b = (const float*)d_in[5];
    const float* ln_w  = (const float*)d_in[6];
    const float* ln_b  = (const float*)d_in[7];
    float* out = (float*)d_out;

    char* ws = (char*)d_ws;
    int* d_idx = (int*)ws;                                       // 32 KB
    const size_t HEAD_HALFS = (size_t)BB * HH * KK * DH;         // 1 M halfs each
    _Float16* d_qf = (_Float16*)(ws + (1 << 16));
    _Float16* d_kf = d_qf + HEAD_HALFS;
    _Float16* d_vf = d_kf + HEAD_HALFS;
    _Float16* d_oh = d_vf + HEAD_HALFS;                          // [B][K][C] f16

    const int n4 = BB * NPB * CC / 4;
    copy_kernel<<<dim3((n4 + 255) / 256), dim3(256), 0, stream>>>(
        (const float4*)feats, (float4*)out, n4);
    topk_kernel<<<dim3(BB), dim3(1024), 0, stream>>>(feats, d_idx);
    qkv_kernel<<<dim3(KK / 64, 6, BB), dim3(128), 0, stream>>>(
        feats, d_idx, in_w, in_b, d_qf, d_kf, d_vf);
    attn_kernel<<<dim3(KK / 64, HH, BB), dim3(128), 0, stream>>>(
        d_qf, d_kf, d_vf, d_oh);
    outproj_kernel<<<dim3(BB * KK), dim3(CC), 0, stream>>>(
        d_oh, feats, d_idx, out_w, out_b, ln_w, ln_b, out);
}
